// XCIT_84645215469743
// MI455X (gfx1250) — matrix operand order repaired and checked by reading, not compiled, not run on hardware
//
#include <hip/hip_runtime.h>
#include <hip/hip_bf16.h>
#include <math.h>

// =====================================================================
// XCiT forward for MI455X (gfx1250).
// GEMM/conv work runs through v_wmma_f32_16x16x32_bf16 (bf16 A/B, f32 acc).
// Block tile 128x128, 8 wave32 waves, wave tile 32x64 -> 8 WMMA / K-step.
// fp32->bf16 via v_cvt_pk_bf16_f32 when available; global_prefetch_b8 on
// the streaming A operand.  Elementwise work is fp32 VALU.
//
// Input flattening assumption (jax tree_flatten = sorted dict keys):
//   0: image
//   1+12*L (L=0,1): cls[L]: fc1.b, fc1.w, fc2.b, fc2.w, ln_a.b, ln_a.g,
//                           ln_m.b, ln_m.g, o.b, o.w, proj.b, proj.w
//   25: cls_token   26: head.b   27: head.w   28: ln_f.b   29: ln_f.g
//   30+6*i (i=0..3): patch[i]: b, be, g, m, v, w
//   54+31*L (L=0..3): xcit[L]: fc1.b, fc1.w, fc2.b, fc2.w, k.b, k.w,
//      ln_lpi.b, ln_lpi.g, ln_mh.b, ln_mh.g, ln_mlp.b, ln_mlp.g,
//      lpi0{b,be,g,m,v,w}, lpi1{b,be,g,m,v,w}, o.b, o.w, q.b, q.w,
//      tau, v.b, v.w
// Spatial: 384 ->(s2,p0,k3) 191 -> 95 -> 47 -> 23 ; N = 529 tokens.
// =====================================================================

typedef __attribute__((ext_vector_type(16))) __bf16 v16bf;
typedef __attribute__((ext_vector_type(8)))  float  v8f;

union FragU { v16bf v; uint4 u[2]; };

__device__ __forceinline__ unsigned short f2bf(float f) {
  unsigned int u = __float_as_uint(f);
  u += 0x7FFFu + ((u >> 16) & 1u);          // round-to-nearest-even
  return (unsigned short)(u >> 16);
}

// pack two fp32 -> packed bf16 (lo = a, hi = b)
#if __has_builtin(__builtin_amdgcn_cvt_pk_bf16_f32)
typedef __attribute__((ext_vector_type(2))) __bf16 v2bf;
__device__ __forceinline__ unsigned int f2bf2(float a, float b) {
  union { v2bf v; unsigned int u; } c;
  c.v = __builtin_amdgcn_cvt_pk_bf16_f32(a, b);
  return c.u;
}
#else
__device__ __forceinline__ unsigned int f2bf2(float a, float b) {
  return (unsigned int)f2bf(a) | ((unsigned int)f2bf(b) << 16);
}
#endif

__device__ __forceinline__ float gelu_f(float x) {
  return 0.5f * x * (1.0f + erff(x * 0.7071067811865476f));
}

// Fragment loads per CDNA5 ISA 16-bit layouts (CDNA5 ISA):
// A 16x32: lane l (m=l&15, half=l>>4) holds K = half*8 + {0..7} and 16+half*8+{0..7}
// B 32x16: stored transposed in LDS [n][k]; lane l (n=l&15, kh=l>>4) holds K = kh*8 + {0..7} and 16+kh*8+{0..7}
__device__ __forceinline__ v16bf frag_a(const unsigned short* As, int tm, int lane) {
  int m = tm + (lane & 15), half = lane >> 4;
  const unsigned short* row = As + m * 40;
  FragU f;
  f.u[0] = *(const uint4*)(row + half * 8);
  f.u[1] = *(const uint4*)(row + 16 + half * 8);
  return f.v;
}
__device__ __forceinline__ v16bf frag_b(const unsigned short* Bs, int tn, int lane) {
  int n = tn + (lane & 15), kh = lane >> 4;
  const unsigned short* row = Bs + n * 40;
  FragU f;
  f.u[0] = *(const uint4*)(row + kh * 8);
  f.u[1] = *(const uint4*)(row + 16 + kh * 8);
  return f.v;
}

#define WMMA_BF16(A, B, C) \
  __builtin_amdgcn_wmma_f32_16x16x32_bf16(false, (A), false, (B), (short)0, (C), false, false)

// ---------------------------------------------------------------------
// Generic GEMM: C[M,N] = act(A[M,K] * Bw[K,N] + bias) (+Res).  K % 32 == 0,
// N % 8 == 0.  A fp32 row-major, Bw bf16 row-major.  256 thr, tile 128x128.
// ---------------------------------------------------------------------
__global__ void __launch_bounds__(256) gemm_wmma(
    const float* __restrict__ A, const unsigned short* __restrict__ Bw,
    const float* __restrict__ bias, const float* __restrict__ Res,
    float* __restrict__ C, int M, int N, int K, int act) {
  __shared__ unsigned short As[128 * 40];
  __shared__ unsigned short Bs[128 * 40];
  const int tid = threadIdx.x;
  const int m0 = blockIdx.y * 128, n0 = blockIdx.x * 128;
  const int lane = tid & 31, w = tid >> 5;
  const int wm = (w & 3) * 32, wn = (w >> 2) * 64;
  v8f acc[2][4] = {};
  const int ac = (tid & 7) * 4, ar0 = tid >> 3;      // A loader
  const int bk = tid & 31, bn = (tid >> 5) * 8;       // B loader

  for (int k0 = 0; k0 < K; k0 += 32) {
    for (int i = 0; i < 4; i++) {
      int r = ar0 + i * 32, gm = m0 + r;
      float4 f = make_float4(0.f, 0.f, 0.f, 0.f);
      if (gm < M) {
        f = *(const float4*)(A + (size_t)gm * K + k0 + ac);
        if (k0 + 32 < K) __builtin_prefetch(A + (size_t)gm * K + k0 + 32 + ac, 0, 0);
      }
      unsigned int* d = (unsigned int*)(As + r * 40 + ac);
      d[0] = f2bf2(f.x, f.y);
      d[1] = f2bf2(f.z, f.w);
    }
    for (int j2 = 0; j2 < 128; j2 += 64) {
      int gn = n0 + bn + j2;
      union { uint4 q; unsigned short s[8]; } tb;
      tb.q = make_uint4(0u, 0u, 0u, 0u);
      if (gn < N) tb.q = *(const uint4*)(Bw + (size_t)(k0 + bk) * N + gn);
      for (int j = 0; j < 8; j++) Bs[(bn + j2 + j) * 40 + bk] = tb.s[j];
    }
    __syncthreads();
    v16bf a0 = frag_a(As, wm, lane), a1 = frag_a(As, wm + 16, lane);
    v16bf b0 = frag_b(Bs, wn, lane),      b1 = frag_b(Bs, wn + 16, lane);
    v16bf b2 = frag_b(Bs, wn + 32, lane), b3 = frag_b(Bs, wn + 48, lane);
    acc[0][0] = WMMA_BF16(a0, b0, acc[0][0]);
    acc[0][1] = WMMA_BF16(a0, b1, acc[0][1]);
    acc[0][2] = WMMA_BF16(a0, b2, acc[0][2]);
    acc[0][3] = WMMA_BF16(a0, b3, acc[0][3]);
    acc[1][0] = WMMA_BF16(a1, b0, acc[1][0]);
    acc[1][1] = WMMA_BF16(a1, b1, acc[1][1]);
    acc[1][2] = WMMA_BF16(a1, b2, acc[1][2]);
    acc[1][3] = WMMA_BF16(a1, b3, acc[1][3]);
    __syncthreads();
  }
  const int half = lane >> 4, nl = lane & 15;
  for (int i = 0; i < 2; i++)
    for (int j = 0; j < 4; j++)
      for (int r = 0; r < 8; r++) {
        int gm = m0 + wm + i * 16 + r + 8 * half;
        int gn = n0 + wn + j * 16 + nl;
        if (gm < M && gn < N) {
          float v = acc[i][j][r];
          if (bias) v += bias[gn];
          if (act == 1) v = gelu_f(v);
          if (Res) v += Res[(size_t)gm * N + gn];
          C[(size_t)gm * N + gn] = v;
        }
      }
}

// ---------------------------------------------------------------------
// Implicit-GEMM 3x3 conv + BN (+GELU).  In: NCHW fp32.  Wt: bf16 [Kpad,Cout]
// (k = ci*9 + kh*3 + kw).  Out: NCHW fp32.  Cout % 8 == 0, Kpad % 32 == 0.
// ---------------------------------------------------------------------
__global__ void __launch_bounds__(256) conv_wmma(
    const float* __restrict__ In, const unsigned short* __restrict__ Wt,
    const float* __restrict__ cb, const float* __restrict__ cg,
    const float* __restrict__ cbe, const float* __restrict__ cm,
    const float* __restrict__ cv, float* __restrict__ Out,
    int M, int Cin, int IH, int IW, int Cout, int OH, int OW,
    int stride, int pad, int Kreal, int Kpad, int act) {
  __shared__ unsigned short As[128 * 40];
  __shared__ unsigned short Bs[128 * 40];
  const int tid = threadIdx.x;
  const int m0 = blockIdx.y * 128, n0 = blockIdx.x * 128;
  const int lane = tid & 31, w = tid >> 5;
  const int wm = (w & 3) * 32, wn = (w >> 2) * 64;
  const int OHW = OH * OW;
  v8f acc[2][4] = {};
  const int ac = (tid & 7) * 4, ar0 = tid >> 3;
  const int bk = tid & 31, bn = (tid >> 5) * 8;

  for (int k0 = 0; k0 < Kpad; k0 += 32) {
    for (int i = 0; i < 4; i++) {
      int r = ar0 + i * 32, gm = m0 + r;
      int b = 0, oy = 0, ox = 0;
      bool rv = gm < M;
      if (rv) { b = gm / OHW; int rem = gm - b * OHW; oy = rem / OW; ox = rem - oy * OW; }
      float val[4];
      for (int j = 0; j < 4; j++) {
        int k = k0 + ac + j;
        val[j] = 0.f;
        if (rv && k < Kreal) {
          int ci = k / 9, kr = k - ci * 9, kh = kr / 3, kw = kr - kh * 3;
          int iy = oy * stride - pad + kh, ix = ox * stride - pad + kw;
          if (iy >= 0 && iy < IH && ix >= 0 && ix < IW)
            val[j] = In[(((size_t)b * Cin + ci) * IH + iy) * IW + ix];
        }
      }
      unsigned int* d = (unsigned int*)(As + r * 40 + ac);
      d[0] = f2bf2(val[0], val[1]);
      d[1] = f2bf2(val[2], val[3]);
    }
    for (int j2 = 0; j2 < 128; j2 += 64) {
      int gn = n0 + bn + j2;
      union { uint4 q; unsigned short s[8]; } tb;
      tb.q = make_uint4(0u, 0u, 0u, 0u);
      if (gn < Cout) tb.q = *(const uint4*)(Wt + (size_t)(k0 + bk) * Cout + gn);
      for (int j = 0; j < 8; j++) Bs[(bn + j2 + j) * 40 + bk] = tb.s[j];
    }
    __syncthreads();
    v16bf a0 = frag_a(As, wm, lane), a1 = frag_a(As, wm + 16, lane);
    v16bf b0 = frag_b(Bs, wn, lane),      b1 = frag_b(Bs, wn + 16, lane);
    v16bf b2 = frag_b(Bs, wn + 32, lane), b3 = frag_b(Bs, wn + 48, lane);
    acc[0][0] = WMMA_BF16(a0, b0, acc[0][0]);
    acc[0][1] = WMMA_BF16(a0, b1, acc[0][1]);
    acc[0][2] = WMMA_BF16(a0, b2, acc[0][2]);
    acc[0][3] = WMMA_BF16(a0, b3, acc[0][3]);
    acc[1][0] = WMMA_BF16(a1, b0, acc[1][0]);
    acc[1][1] = WMMA_BF16(a1, b1, acc[1][1]);
    acc[1][2] = WMMA_BF16(a1, b2, acc[1][2]);
    acc[1][3] = WMMA_BF16(a1, b3, acc[1][3]);
    __syncthreads();
  }
  const int half = lane >> 4, nl = lane & 15;
  for (int i = 0; i < 2; i++)
    for (int j = 0; j < 4; j++)
      for (int r = 0; r < 8; r++) {
        int gm = m0 + wm + i * 16 + r + 8 * half;
        int co = n0 + wn + j * 16 + nl;
        if (gm < M && co < Cout) {
          float scale = cg[co] * rsqrtf(cv[co] + 1e-5f);
          float val = (acc[i][j][r] + cb[co] - cm[co]) * scale + cbe[co];
          if (act == 1) val = gelu_f(val);
          int b = gm / OHW, rem = gm - b * OHW;
          Out[((size_t)b * Cout + co) * OHW + rem] = val;
        }
      }
}

// --------------------------- weight packing ---------------------------
__global__ void pack_bf16(const float* __restrict__ w, unsigned short* __restrict__ o, size_t n) {
  size_t i = (size_t)blockIdx.x * blockDim.x + threadIdx.x;
  if (i < n) o[i] = f2bf(w[i]);
}
__global__ void pack_conv_wt(const float* __restrict__ w, unsigned short* __restrict__ o,
                             int Cout, int Kreal, int Kpad) {
  size_t i = (size_t)blockIdx.x * blockDim.x + threadIdx.x;
  size_t tot = (size_t)Kpad * Cout;
  if (i >= tot) return;
  int k = (int)(i / Cout), n = (int)(i - (size_t)k * Cout);
  o[i] = (k < Kreal) ? f2bf(w[(size_t)n * Kreal + k]) : (unsigned short)0;
}

// ------------------------- elementwise kernels ------------------------
__global__ void layernorm_k(const float* __restrict__ x, const float* __restrict__ g,
                            const float* __restrict__ b, float* __restrict__ y) {
  const int row = blockIdx.x, t = threadIdx.x;           // 256 threads, E=768
  __shared__ float red[256];
  const float* xr = x + (size_t)row * 768;
  float s = 0.f;
  for (int i = t; i < 768; i += 256) s += xr[i];
  red[t] = s; __syncthreads();
  for (int o = 128; o > 0; o >>= 1) { if (t < o) red[t] += red[t + o]; __syncthreads(); }
  float mu = red[0] * (1.f / 768.f);
  __syncthreads();
  float v = 0.f;
  for (int i = t; i < 768; i += 256) { float d = xr[i] - mu; v += d * d; }
  red[t] = v; __syncthreads();
  for (int o = 128; o > 0; o >>= 1) { if (t < o) red[t] += red[t + o]; __syncthreads(); }
  float inv = rsqrtf(red[0] * (1.f / 768.f) + 1e-5f);
  float* yr = y + (size_t)row * 768;
  for (int i = t; i < 768; i += 256) yr[i] = (xr[i] - mu) * inv * g[i] + b[i];
}

// L2-normalize over token axis: data [B,N,768]; one block per (b, channel)
__global__ void l2norm_tokens(float* __restrict__ q, int N) {
  const int b = blockIdx.x / 768, e = blockIdx.x % 768, t = threadIdx.x; // 64 thr
  float* base = q + (size_t)b * N * 768 + e;
  __shared__ float red[64];
  float s = 0.f;
  for (int n = t; n < N; n += 64) { float v = base[(size_t)n * 768]; s += v * v; }
  red[t] = s; __syncthreads();
  for (int o = 32; o > 0; o >>= 1) { if (t < o) red[t] += red[t + o]; __syncthreads(); }
  float scale = 1.f / fmaxf(sqrtf(red[0]), 1e-12f);
  for (int n = t; n < N; n += 64) base[(size_t)n * 768] *= scale;
}

// energy[b,e,h,q] = sum_n q[b,n,h,e] * k[b,n,h,q]
__global__ void xca_energy(const float* __restrict__ q, const float* __restrict__ k,
                           float* __restrict__ en, int N) {
  int idx = blockIdx.x * blockDim.x + threadIdx.x;
  if (idx >= 16 * 96 * 8 * 96) return;
  int qq = idx % 96, h = (idx / 96) % 8, e = (idx / 768) % 96, b = idx / 73728;
  const float* qp = q + (size_t)b * N * 768 + h * 96 + e;
  const float* kp = k + (size_t)b * N * 768 + h * 96 + qq;
  float s = 0.f;
  for (int n = 0; n < N; n++) s += qp[(size_t)n * 768] * kp[(size_t)n * 768];
  en[idx] = s;
}

// softmax over last axis (96), then divide column q by tau[q]; row = (b,e,h)
__global__ void softmax_tau(float* __restrict__ en, const float* __restrict__ tau, int rows) {
  int r = blockIdx.x * blockDim.x + threadIdx.x;
  if (r >= rows) return;
  float* p = en + (size_t)r * 96;
  float mx = -1e30f;
  for (int i = 0; i < 96; i++) mx = fmaxf(mx, p[i]);
  float s = 0.f;
  for (int i = 0; i < 96; i++) { float e = expf(p[i] - mx); p[i] = e; s += e; }
  float inv = 1.f / s;
  for (int i = 0; i < 96; i++) p[i] = p[i] * inv / tau[i];
}

// out[b,n,h,q] = sum_e a[b,e,h,q] * v[b,n,h,e]
__global__ void xca_apply(const float* __restrict__ a, const float* __restrict__ v,
                          float* __restrict__ o, int N) {
  int idx = blockIdx.x * blockDim.x + threadIdx.x;
  if (idx >= 16 * N * 768) return;
  int qq = idx % 96, h = (idx / 96) % 8, n = (idx / 768) % N, b = idx / (768 * N);
  const float* vp = v + ((size_t)b * N + n) * 768 + h * 96;
  const float* ap = a + (size_t)b * 73728 + h * 96 + qq;
  float s = 0.f;
  for (int e = 0; e < 96; e++) s += ap[(size_t)e * 768] * vp[e];
  o[idx] = s;
}

__global__ void btc_to_bct(const float* __restrict__ in, float* __restrict__ out, int N) {
  int idx = blockIdx.x * blockDim.x + threadIdx.x;
  if (idx >= 16 * N * 768) return;
  int c = idx % 768, n = (idx / 768) % N, b = idx / (768 * N);
  out[((size_t)b * 768 + c) * N + n] = in[idx];
}
__global__ void bct_to_btc(const float* __restrict__ in, float* __restrict__ out, int N) {
  int idx = blockIdx.x * blockDim.x + threadIdx.x;
  if (idx >= 16 * N * 768) return;
  int c = idx % 768, n = (idx / 768) % N, b = idx / (768 * N);
  out[idx] = in[((size_t)b * 768 + c) * N + n];
}
__global__ void bct_to_btc_add(const float* __restrict__ in, float* __restrict__ out, int N) {
  int idx = blockIdx.x * blockDim.x + threadIdx.x;
  if (idx >= 16 * N * 768) return;
  int c = idx % 768, n = (idx / 768) % N, b = idx / (768 * N);
  out[idx] += in[((size_t)b * 768 + c) * N + n];
}

// ------------------------- class-attention stage ----------------------
__global__ void build_tok(const float* __restrict__ x, const float* __restrict__ clst,
                          float* __restrict__ tok) {
  int idx = blockIdx.x * blockDim.x + threadIdx.x;
  if (idx >= 16 * 530 * 768) return;
  int e = idx % 768, n = (idx / 768) % 530, b = idx / (530 * 768);
  tok[idx] = (n == 0) ? clst[e] : x[((size_t)b * 529 + (n - 1)) * 768 + e];
}
// attn[b,n,h] = <q0, k_n> / sqrt(96);  proj = [B,530, q(768)|k(768)|v(768)]
__global__ void cls_scores(const float* __restrict__ proj, float* __restrict__ attn) {
  int idx = blockIdx.x * blockDim.x + threadIdx.x;
  if (idx >= 16 * 530 * 8) return;
  int h = idx % 8, n = (idx / 8) % 530, b = idx / (8 * 530);
  const float* qp = proj + (size_t)b * 530 * 2304 + h * 96;
  const float* kp = proj + ((size_t)b * 530 + n) * 2304 + 768 + h * 96;
  float s = 0.f;
  for (int d = 0; d < 96; d++) s += qp[d] * kp[d];
  attn[idx] = s * 0.10206207261596575f;
}
__global__ void cls_softmax_k(float* __restrict__ attn) {        // over n, per (b,h)
  int idx = blockIdx.x * blockDim.x + threadIdx.x;
  if (idx >= 16 * 8) return;
  int h = idx & 7, b = idx >> 3;
  float* p = attn + (size_t)b * 530 * 8 + h;
  float mx = -1e30f;
  for (int n = 0; n < 530; n++) mx = fmaxf(mx, p[(size_t)n * 8]);
  float s = 0.f;
  for (int n = 0; n < 530; n++) { float e = expf(p[(size_t)n * 8] - mx); p[(size_t)n * 8] = e; s += e; }
  float inv = 1.f / s;
  for (int n = 0; n < 530; n++) p[(size_t)n * 8] *= inv;
}
__global__ void cls_combine(const float* __restrict__ attn, const float* __restrict__ proj,
                            float* __restrict__ tokattn) {
  int idx = blockIdx.x * blockDim.x + threadIdx.x;
  if (idx >= 16 * 768) return;
  int d = idx % 96, h = (idx / 96) % 8, b = idx / 768;
  float s = 0.f;
  for (int n = 0; n < 530; n++)
    s += attn[((size_t)b * 530 + n) * 8 + h] *
         proj[((size_t)b * 530 + n) * 2304 + 1536 + h * 96 + d];
  tokattn[idx] = s;
}
__global__ void cls_attn_residual(float* __restrict__ tok, const float* __restrict__ z,
                                  const float* __restrict__ obuf) {
  int idx = blockIdx.x * blockDim.x + threadIdx.x;
  if (idx >= 16 * 530 * 768) return;
  int e = idx % 768, n = (idx / 768) % 530, b = idx / (530 * 768);
  tok[idx] += (n == 0) ? obuf[(size_t)b * 768 + e] : z[idx];
}
__global__ void extract_row0(const float* __restrict__ x, float* __restrict__ out, int N) {
  int idx = blockIdx.x * blockDim.x + threadIdx.x;
  if (idx >= 16 * 768) return;
  out[idx] = x[(size_t)(idx / 768) * N * 768 + (idx % 768)];
}
// final = x + cat(cls+mlp(cls), x[:,1:])  (m2 already = cls + mlp(cls))
__global__ void finish_cls(float* __restrict__ tok, const float* __restrict__ m2) {
  int idx = blockIdx.x * blockDim.x + threadIdx.x;
  if (idx >= 16 * 530 * 768) return;
  int e = idx % 768, n = (idx / 768) % 530, b = idx / (530 * 768);
  tok[idx] = (n == 0) ? (tok[idx] + m2[(size_t)b * 768 + e]) : (2.f * tok[idx]);
}

// =====================================================================
extern "C" void kernel_launch(void* const* d_in, const int* in_sizes, int n_in,
                              void* d_out, int out_size, void* d_ws, size_t ws_size,
                              hipStream_t stream) {
  (void)in_sizes; (void)n_in; (void)out_size;
  auto F = [&](int i) -> const float* { return (const float*)d_in[i]; };
  const float* image = F(0);

  // -------- workspace bump arena --------
  char* wbase = (char*)d_ws;
  size_t off = 0;
  auto alloc = [&](size_t bytes) -> void* {
    void* p = wbase + off;
    off = (off + bytes + 255) & ~(size_t)255;
    return p;
  };

  const int NT = 529, NT1 = 530;
  const int MT = 16 * NT;      // 8464 token rows
  const int MT1 = 16 * NT1;    // 8480 (with cls token)

  const int pCin[4] = {3, 96, 192, 384}, pCout[4] = {96, 192, 384, 768};
  const int pIH[4] = {384, 191, 95, 47}, pOH[4] = {191, 95, 47, 23};
  int pKreal[4], pKpad[4];
  unsigned short* WtPatch[4];
  for (int i = 0; i < 4; i++) {
    pKreal[i] = pCin[i] * 9;
    pKpad[i] = (pKreal[i] + 31) & ~31;
    WtPatch[i] = (unsigned short*)alloc((size_t)pKpad[i] * pCout[i] * 2);
  }
  unsigned short* WtLpi[4][2];
  unsigned short *Wq[4], *Wk[4], *Wv[4], *Wo[4], *Wf1[4], *Wf2[4];
  for (int L = 0; L < 4; L++) {
    WtLpi[L][0] = (unsigned short*)alloc((size_t)6912 * 768 * 2);
    WtLpi[L][1] = (unsigned short*)alloc((size_t)6912 * 768 * 2);
    Wq[L]  = (unsigned short*)alloc((size_t)768 * 768 * 2);
    Wk[L]  = (unsigned short*)alloc((size_t)768 * 768 * 2);
    Wv[L]  = (unsigned short*)alloc((size_t)768 * 768 * 2);
    Wo[L]  = (unsigned short*)alloc((size_t)768 * 768 * 2);
    Wf1[L] = (unsigned short*)alloc((size_t)768 * 3072 * 2);
    Wf2[L] = (unsigned short*)alloc((size_t)3072 * 768 * 2);
  }
  unsigned short *Wproj[2], *Woc[2], *Wf1c[2], *Wf2c[2];
  for (int L = 0; L < 2; L++) {
    Wproj[L] = (unsigned short*)alloc((size_t)768 * 2304 * 2);
    Woc[L]   = (unsigned short*)alloc((size_t)768 * 768 * 2);
    Wf1c[L]  = (unsigned short*)alloc((size_t)768 * 3072 * 2);
    Wf2c[L]  = (unsigned short*)alloc((size_t)3072 * 768 * 2);
  }
  unsigned short* Whead = (unsigned short*)alloc((size_t)768 * 1000 * 2);

  float* big0 = (float*)alloc((size_t)16 * 96 * 191 * 191 * 4);  // conv1 out; reused: mlp-hidden / cls-proj
  float* big1 = (float*)alloc((size_t)16 * 192 * 95 * 95 * 4);   // conv2 out; reused: conv4 out
  size_t xbytes = (size_t)MT * 768 * 4;
  float* x   = (float*)alloc(xbytes);
  float* lnb = (float*)alloc(xbytes);
  float* qb  = (float*)alloc(xbytes);
  float* kb  = (float*)alloc(xbytes);
  float* vb  = (float*)alloc(xbytes);
  float* ao  = (float*)alloc(xbytes);
  float* energy = (float*)alloc((size_t)16 * 96 * 8 * 96 * 4);
  float* t1 = (float*)alloc(xbytes);
  float* t2 = (float*)alloc(xbytes);
  size_t tokbytes = (size_t)MT1 * 768 * 4;
  float* tok   = (float*)alloc(tokbytes);
  float* ztok  = (float*)alloc(tokbytes);
  float* attnc = (float*)alloc((size_t)MT1 * 8 * 4);
  float* tokattn = (float*)alloc((size_t)16 * 768 * 4);
  float* obuf = (float*)alloc((size_t)16 * 768 * 4);
  float* cbuf = (float*)alloc((size_t)16 * 768 * 4);
  float* m1c  = (float*)alloc((size_t)16 * 3072 * 4);
  float* m2   = (float*)alloc((size_t)16 * 768 * 4);
  float* t0   = (float*)alloc((size_t)16 * 768 * 4);
  if (off > ws_size) return;  // insufficient scratch: refuse to scribble OOB

  // -------- launch helpers --------
  auto packLin = [&](const float* w, unsigned short* dst, size_t n) {
    pack_bf16<<<dim3((unsigned)((n + 255) / 256)), dim3(256), 0, stream>>>(w, dst, n);
  };
  auto packConv = [&](const float* w, unsigned short* dst, int Cout, int Kreal, int Kpad) {
    size_t tot = (size_t)Kpad * Cout;
    pack_conv_wt<<<dim3((unsigned)((tot + 255) / 256)), dim3(256), 0, stream>>>(w, dst, Cout, Kreal, Kpad);
  };
  auto gemm = [&](const float* A, const unsigned short* Bw, const float* bias,
                  const float* Res, float* C, int M, int N, int K, int act) {
    dim3 g((N + 127) / 128, (M + 127) / 128);
    gemm_wmma<<<g, 256, 0, stream>>>(A, Bw, bias, Res, C, M, N, K, act);
  };
  auto convL = [&](const float* In, const unsigned short* Wt, const float* b_,
                   const float* g_, const float* be_, const float* m_, const float* v_,
                   float* Out, int Cin, int IH, int Cout, int OH,
                   int stride, int pad, int Kreal, int Kpad, int act) {
    int M = 16 * OH * OH;
    dim3 g((Cout + 127) / 128, (M + 127) / 128);
    conv_wmma<<<g, 256, 0, stream>>>(In, Wt, b_, g_, be_, m_, v_, Out, M, Cin, IH, IH,
                                     Cout, OH, OH, stride, pad, Kreal, Kpad, act);
  };
  auto ew = [&](int total) { return dim3((total + 255) / 256); };

  // -------- pack all weights to bf16 --------
  for (int i = 0; i < 4; i++) packConv(F(30 + 6 * i + 5), WtPatch[i], pCout[i], pKreal[i], pKpad[i]);
  for (int L = 0; L < 4; L++) {
    int xb = 54 + 31 * L;
    packConv(F(xb + 17), WtLpi[L][0], 768, 6912, 6912);
    packConv(F(xb + 23), WtLpi[L][1], 768, 6912, 6912);
    packLin(F(xb + 27), Wq[L],  (size_t)768 * 768);
    packLin(F(xb + 5),  Wk[L],  (size_t)768 * 768);
    packLin(F(xb + 30), Wv[L],  (size_t)768 * 768);
    packLin(F(xb + 25), Wo[L],  (size_t)768 * 768);
    packLin(F(xb + 1),  Wf1[L], (size_t)768 * 3072);
    packLin(F(xb + 3),  Wf2[L], (size_t)3072 * 768);
  }
  for (int L = 0; L < 2; L++) {
    int cb = 1 + 12 * L;
    packLin(F(cb + 11), Wproj[L], (size_t)768 * 2304);
    packLin(F(cb + 9),  Woc[L],   (size_t)768 * 768);
    packLin(F(cb + 1),  Wf1c[L],  (size_t)768 * 3072);
    packLin(F(cb + 3),  Wf2c[L],  (size_t)3072 * 768);
  }
  packLin(F(27), Whead, (size_t)768 * 1000);

  // -------- patch embedding: 4x conv(s2,p0) (GELU on first three) --------
  float* pbuf[4] = {big0, big1, big0, big1};
  const float* pin = image;
  for (int i = 0; i < 4; i++) {
    int pb = 30 + 6 * i;
    convL(pin, WtPatch[i], F(pb + 0), F(pb + 2), F(pb + 1), F(pb + 3), F(pb + 4),
          pbuf[i], pCin[i], pIH[i], pCout[i], pOH[i], 2, 0, pKreal[i], pKpad[i],
          (i < 3) ? 1 : 0);
    pin = pbuf[i];
  }
  bct_to_btc<<<ew(16 * NT * 768), 256, 0, stream>>>(big1, x, NT);  // [B,768,529] -> [B,529,768]

  // -------- 4 XCiT layers --------
  for (int L = 0; L < 4; L++) {
    int xb = 54 + 31 * L;
    // XCA block
    layernorm_k<<<MT, 256, 0, stream>>>(x, F(xb + 9), F(xb + 8), lnb);    // ln_mh
    gemm(lnb, Wq[L], F(xb + 26), nullptr, qb, MT, 768, 768, 0);
    gemm(lnb, Wk[L], F(xb + 4),  nullptr, kb, MT, 768, 768, 0);
    gemm(lnb, Wv[L], F(xb + 29), nullptr, vb, MT, 768, 768, 0);
    l2norm_tokens<<<16 * 768, 64, 0, stream>>>(qb, NT);
    l2norm_tokens<<<16 * 768, 64, 0, stream>>>(kb, NT);
    xca_energy<<<ew(16 * 96 * 8 * 96), 256, 0, stream>>>(qb, kb, energy, NT);
    softmax_tau<<<ew(16 * 96 * 8), 256, 0, stream>>>(energy, F(xb + 28), 16 * 96 * 8);
    xca_apply<<<ew(16 * NT * 768), 256, 0, stream>>>(energy, vb, ao, NT);
    gemm(ao, Wo[L], F(xb + 24), x, x, MT, 768, 768, 0);                   // x += o(attn)
    // LPI block
    layernorm_k<<<MT, 256, 0, stream>>>(x, F(xb + 7), F(xb + 6), lnb);    // ln_lpi
    btc_to_bct<<<ew(16 * NT * 768), 256, 0, stream>>>(lnb, t1, NT);
    convL(t1, WtLpi[L][0], F(xb + 12), F(xb + 14), F(xb + 13), F(xb + 15), F(xb + 16),
          t2, 768, 23, 768, 23, 1, 1, 6912, 6912, 1);
    convL(t2, WtLpi[L][1], F(xb + 18), F(xb + 20), F(xb + 19), F(xb + 21), F(xb + 22),
          t1, 768, 23, 768, 23, 1, 1, 6912, 6912, 1);
    bct_to_btc_add<<<ew(16 * NT * 768), 256, 0, stream>>>(t1, x, NT);     // x += lpi
    // MLP block
    layernorm_k<<<MT, 256, 0, stream>>>(x, F(xb + 11), F(xb + 10), lnb);  // ln_mlp
    gemm(lnb, Wf1[L], F(xb + 0), nullptr, big0, MT, 3072, 768, 1);        // gelu
    gemm(big0, Wf2[L], F(xb + 2), x, x, MT, 768, 3072, 0);                // x += mlp
  }

  // -------- class-attention stage --------
  build_tok<<<ew(16 * NT1 * 768), 256, 0, stream>>>(x, F(25), tok);
  for (int L = 0; L < 2; L++) {
    int cb = 1 + 12 * L;
    layernorm_k<<<MT1, 256, 0, stream>>>(tok, F(cb + 5), F(cb + 4), ztok);     // ln_a
    gemm(ztok, Wproj[L], F(cb + 10), nullptr, big0, MT1, 2304, 768, 0);        // qkv
    cls_scores<<<ew(16 * NT1 * 8), 256, 0, stream>>>(big0, attnc);
    cls_softmax_k<<<1, 128, 0, stream>>>(attnc);
    cls_combine<<<ew(16 * 768), 256, 0, stream>>>(attnc, big0, tokattn);
    gemm(tokattn, Woc[L], F(cb + 8), nullptr, obuf, 16, 768, 768, 0);
    cls_attn_residual<<<ew(16 * NT1 * 768), 256, 0, stream>>>(tok, ztok, obuf);
    layernorm_k<<<MT1, 256, 0, stream>>>(tok, F(cb + 7), F(cb + 6), tok);      // ln_m (in-place)
    extract_row0<<<ew(16 * 768), 256, 0, stream>>>(tok, cbuf, NT1);
    gemm(cbuf, Wf1c[L], F(cb + 0), nullptr, m1c, 16, 3072, 768, 1);
    gemm(m1c, Wf2c[L], F(cb + 2), cbuf, m2, 16, 768, 3072, 0);                 // m2 = cls + mlp(cls)
    finish_cls<<<ew(16 * NT1 * 768), 256, 0, stream>>>(tok, m2);
  }

  // -------- final LN on cls token + head --------
  extract_row0<<<ew(16 * 768), 256, 0, stream>>>(tok, t0, NT1);
  layernorm_k<<<16, 256, 0, stream>>>(t0, F(29), F(28), t0);
  gemm(t0, Whead, F(26), nullptr, (float*)d_out, 16, 1000, 768, 0);
}
